// SelfGNN_46351287058918
// MI455X (gfx1250) — matrix operand order repaired and checked by reading, not compiled, not run on hardware
//
#include <hip/hip_runtime.h>
#include <hip/hip_bf16.h>

typedef __bf16 bf16t;
typedef __attribute__((ext_vector_type(16))) __bf16 v16bf;
typedef __attribute__((ext_vector_type(8)))  __bf16 v8bf;
typedef __attribute__((ext_vector_type(8)))  float  v8f;

#define BQ 2048
#define LQ 50
#define DQ 128
#define NVQ 50001

// ---------------------------------------------------------------------------
// fp32 -> bf16 conversion
// ---------------------------------------------------------------------------
__global__ void k_f32_to_bf16(const float* __restrict__ in, bf16t* __restrict__ out,
                              size_t n) {
    size_t i = (size_t)blockIdx.x * blockDim.x + threadIdx.x;
    if (i < n) out[i] = (bf16t)in[i];
}

// ---------------------------------------------------------------------------
// x[b,t,:] = item_emb[seq[b,t]] * (seq>0), stored bf16
// ---------------------------------------------------------------------------
__global__ void k_gather_x(const int* __restrict__ seq, const float* __restrict__ emb,
                           bf16t* __restrict__ x_bf) {
    size_t i = (size_t)blockIdx.x * blockDim.x + threadIdx.x;
    const size_t total = (size_t)BQ * LQ * DQ;
    if (i >= total) return;
    size_t rd = i >> 7;
    int d = (int)(i & 127);
    int s = seq[rd];
    float v = (s > 0) ? emb[(size_t)s * DQ + d] : 0.0f;
    x_bf[i] = (bf16t)v;
}

// ---------------------------------------------------------------------------
// Generic bf16 WMMA GEMM: C[M,N] = A[M,K] @ W[N,K]^T + bias, optional ReLU.
// 256 threads = 8 waves; each wave owns one 16x16 output tile.
// A tile (16 x K) staged in LDS; W fragments read directly (hot in L2).
// K must be a multiple of 32 (here 128 or 512). M multiple of 16.
// ---------------------------------------------------------------------------
__global__ __launch_bounds__(256)
void k_gemm_bf16(const bf16t* __restrict__ A, const bf16t* __restrict__ W,
                 const float* __restrict__ bias,
                 float* __restrict__ outF, bf16t* __restrict__ outB,
                 int M, int N, int K, int relu) {
    __shared__ __align__(32) bf16t sA[16 * 512];
    const int tid  = threadIdx.x;
    const int lane = tid & 31;
    const int wave = tid >> 5;
    const int half = lane >> 4;
    const int l16  = lane & 15;
    const size_t tileM = (size_t)blockIdx.y * 16;

    for (int i = tid; i < 16 * K; i += 256) {
        int r = i / K, c = i - r * K;
        sA[i] = A[(tileM + r) * (size_t)K + c];
    }
    __syncthreads();

    const int tileN = (blockIdx.x * 8 + wave) * 16;
    if (tileN >= N) return;

    int wn = tileN + l16;
    if (wn > N - 1) wn = N - 1;                 // clamp for ragged last tile
    const bf16t* wrow = W + (size_t)wn * K;

    v8f c = {};
    for (int k0 = 0; k0 < K; k0 += 32) {
        union { v16bf v; v8bf h[2]; } a;
        a.h[0] = *(const v8bf*)(&sA[l16 * K + k0 + half * 8]);
        a.h[1] = *(const v8bf*)(&sA[l16 * K + k0 + 16 + half * 8]);
        union { v16bf v; v8bf h[2]; } bu;
        bu.h[0] = *(const v8bf*)(wrow + k0 + half * 8);
        bu.h[1] = *(const v8bf*)(wrow + k0 + 16 + half * 8);
        v16bf bfrag = bu.v;
        c = __builtin_amdgcn_wmma_f32_16x16x32_bf16(false, a.v, false, bfrag,
                                                    (short)0, c, false, false);
    }

    const int col = tileN + l16;
    if (col < N) {
        const float bv = bias ? bias[col] : 0.0f;
        for (int v = 0; v < 8; ++v) {
            size_t row = tileM + (size_t)(half * 8 + v);
            float val = c[v] + bv;
            if (relu) val = fmaxf(val, 0.0f);
            if (outF) outF[row * (size_t)N + col] = val;
            if (outB) outB[row * (size_t)N + col] = (bf16t)val;
        }
    }
}

// ---------------------------------------------------------------------------
// GRU scan: 16 batch rows per block, WMMA for gh = h @ Whh^T each timestep.
// gi already holds x@Wih^T + bih (from GEMM). Emits h_seq (f32) and
// (h + pos_emb) as bf16 for the attention qkv GEMM.
// ---------------------------------------------------------------------------
__global__ __launch_bounds__(256)
void k_gru_scan(const float* __restrict__ gi, const bf16t* __restrict__ whh_bf,
                const float* __restrict__ bhh, const float* __restrict__ pos,
                float* __restrict__ h_seq, bf16t* __restrict__ hpos_bf) {
    __shared__ float  sh[16 * 128];
    __shared__ __align__(32) bf16t shb[16 * 128];
    __shared__ float  sgh[16 * 384];
    const int tid  = threadIdx.x;
    const int lane = tid & 31;
    const int wave = tid >> 5;
    const int half = lane >> 4;
    const int l16  = lane & 15;
    const size_t b0 = (size_t)blockIdx.x * 16;

    for (int i = tid; i < 16 * 128; i += 256) { sh[i] = 0.0f; shb[i] = (bf16t)0.0f; }
    __syncthreads();

    for (int t = 0; t < LQ; ++t) {
        // gh tiles: 24 tiles of 16 cols across 8 waves
        for (int tt = wave; tt < 24; tt += 8) {
            const int tileN = tt * 16;
            const bf16t* wrow = whh_bf + (size_t)(tileN + l16) * 128;
            v8f c = {};
            for (int k0 = 0; k0 < 128; k0 += 32) {
                union { v16bf v; v8bf h[2]; } a;
                a.h[0] = *(const v8bf*)(&shb[l16 * 128 + k0 + half * 8]);
                a.h[1] = *(const v8bf*)(&shb[l16 * 128 + k0 + 16 + half * 8]);
                union { v16bf v; v8bf h[2]; } bu;
                bu.h[0] = *(const v8bf*)(wrow + k0 + half * 8);
                bu.h[1] = *(const v8bf*)(wrow + k0 + 16 + half * 8);
                v16bf bfrag = bu.v;
                c = __builtin_amdgcn_wmma_f32_16x16x32_bf16(false, a.v, false, bfrag,
                                                            (short)0, c, false, false);
            }
            for (int v = 0; v < 8; ++v)
                sgh[(half * 8 + v) * 384 + tileN + l16] = c[v];
        }
        __syncthreads();

        for (int i = tid; i < 16 * 128; i += 256) {
            const int r = i >> 7, d = i & 127;
            const size_t row = (b0 + r) * (size_t)LQ + t;
            const float* girow = gi + row * 384;
            float ir  = girow[d];
            float iz  = girow[128 + d];
            float inn = girow[256 + d];
            float hr = sgh[r * 384 + d]       + bhh[d];
            float hz = sgh[r * 384 + 128 + d] + bhh[128 + d];
            float hn = sgh[r * 384 + 256 + d] + bhh[256 + d];
            float rg = 1.0f / (1.0f + __expf(-(ir + hr)));
            float zg = 1.0f / (1.0f + __expf(-(iz + hz)));
            float ng = tanhf(inn + rg * hn);
            float hnew = (1.0f - zg) * ng + zg * sh[i];
            h_seq[row * DQ + d]   = hnew;
            hpos_bf[row * DQ + d] = (bf16t)(hnew + pos[t * DQ + d]);
            sh[i]  = hnew;
            shb[i] = (bf16t)hnew;
        }
        __syncthreads();
    }
}

// ---------------------------------------------------------------------------
// Per-(batch, head) 50x50 attention: softmax(q k^T / 8) v -> o (bf16, heads concat)
// ---------------------------------------------------------------------------
__global__ __launch_bounds__(128)
void k_attention(const float* __restrict__ qkv, bf16t* __restrict__ o_bf) {
    const int b = blockIdx.x, h = blockIdx.y;
    __shared__ float q[50][64], k[50][64], v[50][64], S[50][50];
    const int tid = threadIdx.x;

    for (int i = tid; i < 50 * 64; i += 128) {
        int l = i >> 6, d = i & 63;
        size_t row = ((size_t)b * LQ + l) * 384;
        q[l][d] = qkv[row + h * 64 + d];
        k[l][d] = qkv[row + 128 + h * 64 + d];
        v[l][d] = qkv[row + 256 + h * 64 + d];
    }
    __syncthreads();

    for (int p = tid; p < 2500; p += 128) {
        int i = p / 50, j = p - i * 50;
        float s = 0.0f;
        for (int d = 0; d < 64; ++d) s += q[i][d] * k[j][d];
        S[i][j] = s * 0.125f;
    }
    __syncthreads();

    if (tid < 50) {
        float m = -1e30f;
        for (int j = 0; j < 50; ++j) m = fmaxf(m, S[tid][j]);
        float sum = 0.0f;
        for (int j = 0; j < 50; ++j) { float e = __expf(S[tid][j] - m); S[tid][j] = e; sum += e; }
        float inv = 1.0f / sum;
        for (int j = 0; j < 50; ++j) S[tid][j] *= inv;
    }
    __syncthreads();

    for (int p = tid; p < 50 * 64; p += 128) {
        int i = p >> 6, d = p & 63;
        float s = 0.0f;
        for (int j = 0; j < 50; ++j) s += S[i][j] * v[j][d];
        o_bf[((size_t)b * LQ + i) * DQ + h * 64 + d] = (bf16t)s;
    }
}

// ---------------------------------------------------------------------------
// y = LayerNorm(a + bsrc (+ pos[row%L])) * g + beta ; one row (D=128) per block
// ---------------------------------------------------------------------------
__global__ __launch_bounds__(128)
void k_add_ln(const float* __restrict__ a, const float* __restrict__ bsrc,
              const float* __restrict__ pos, const float* __restrict__ g,
              const float* __restrict__ beta,
              float* __restrict__ outF, bf16t* __restrict__ outB) {
    const size_t row = blockIdx.x;
    const int d = threadIdx.x;
    __shared__ float red[128];

    float x = a[row * DQ + d] + bsrc[row * DQ + d];
    if (pos) x += pos[(row % LQ) * DQ + d];

    red[d] = x; __syncthreads();
    for (int s = 64; s > 0; s >>= 1) { if (d < s) red[d] += red[d + s]; __syncthreads(); }
    const float m = red[0] * (1.0f / 128.0f);
    __syncthreads();

    const float xc = x - m;
    red[d] = xc * xc; __syncthreads();
    for (int s = 64; s > 0; s >>= 1) { if (d < s) red[d] += red[d + s]; __syncthreads(); }
    const float var = red[0] * (1.0f / 128.0f);

    const float y = xc * rsqrtf(var + 1e-5f) * g[d] + beta[d];
    if (outF) outF[row * DQ + d] = y;
    if (outB) outB[row * DQ + d] = (bf16t)y;
}

// ---------------------------------------------------------------------------
// Per-sequence session-graph GNN representation (one block per batch element)
// ---------------------------------------------------------------------------
__global__ __launch_bounds__(128)
void k_gnn(const int* __restrict__ seq, const float* __restrict__ emb,
           float* __restrict__ short_u) {
    const int b = blockIdx.x, tid = threadIdx.x;
    __shared__ int ss[50], inv[50], isfirst[50], uniq[50];
    __shared__ int s_nu;
    __shared__ float A[50][50];
    __shared__ float e[50][128];

    if (tid < 50) ss[tid] = seq[b * LQ + tid];
    __syncthreads();

    if (tid < 50) {
        const int s = ss[tid];
        int fo = 0, cnt = 0;
        if (s > 0) {
            fo = 1;
            for (int j = 0; j < tid; ++j) if (ss[j] == s) { fo = 0; break; }
            for (int j = 0; j < 50; ++j) {
                const int sj = ss[j];
                if (sj > 0 && sj < s) {
                    int first = 1;
                    for (int j2 = 0; j2 < j; ++j2) if (ss[j2] == sj) { first = 0; break; }
                    cnt += first;
                }
            }
        }
        inv[tid] = cnt;
        isfirst[tid] = (s > 0) ? fo : 0;
        uniq[tid] = 0;
    }
    __syncthreads();

    if (tid == 0) {
        int nu = 0;
        for (int j = 0; j < 50; ++j) nu += isfirst[j];
        s_nu = nu;
    }
    __syncthreads();

    if (tid < 50 && isfirst[tid]) uniq[inv[tid]] = ss[tid];
    for (int i = tid; i < 2500; i += 128) A[i / 50][i % 50] = 0.0f;
    __syncthreads();

    if (tid == 0 && s_nu > 1) {
        int prev = -1;
        for (int i = 0; i < 50; ++i) if (ss[i] > 0) {
            if (prev >= 0) A[inv[prev]][inv[i]] = 1.0f;
            prev = i;
        }
    }
    __syncthreads();

    if (tid < 50) {
        float rs = 0.0f;
        for (int j = 0; j < 50; ++j) rs += A[tid][j];
        const float invr = 1.0f / (rs + 1e-8f);
        for (int j = 0; j < 50; ++j) A[tid][j] *= invr;
    }
    __syncthreads();

    const int nu = s_nu;
    for (int i = tid; i < 50 * 128; i += 128) {
        int k = i >> 7, d = i & 127;
        e[k][d] = (k < nu) ? emb[(size_t)uniq[k] * DQ + d] : 0.0f;
    }
    __syncthreads();

    float total = 0.0f;
    for (int k = 0; k < 50; ++k) {
        float acc = 0.0f;
        for (int m = 0; m < 50; ++m) acc += A[k][m] * e[m][tid];
        float lr = (acc > 0.0f) ? acc : 0.01f * acc;
        total += lr + e[k][tid];
    }
    const int den = (nu > 1) ? nu : 1;
    short_u[(size_t)b * DQ + tid] = total / (float)den;
}

// ---------------------------------------------------------------------------
// e_hat = short_u + mean_valid(h_seq) + mean_valid(h_attn), stored bf16
// ---------------------------------------------------------------------------
__global__ __launch_bounds__(128)
void k_pool(const int* __restrict__ seq, const float* __restrict__ h_seq,
            const float* __restrict__ h_attn, const float* __restrict__ short_u,
            bf16t* __restrict__ ehat_bf) {
    const int b = blockIdx.x, d = threadIdx.x;
    float sb = 0.0f, st = 0.0f, cnt = 0.0f;
    for (int t = 0; t < LQ; ++t) {
        const float m = (seq[b * LQ + t] > 0) ? 1.0f : 0.0f;
        cnt += m;
        const size_t row = ((size_t)b * LQ + t) * DQ + d;
        sb += m * h_seq[row];
        st += m * h_attn[row];
    }
    const float len = fmaxf(cnt, 1.0f);
    const float e = short_u[(size_t)b * DQ + d] + sb / len + st / len;
    ehat_bf[(size_t)b * DQ + d] = (bf16t)e;
}

// ---------------------------------------------------------------------------
// Host launcher
// ---------------------------------------------------------------------------
extern "C" void kernel_launch(void* const* d_in, const int* in_sizes, int n_in,
                              void* d_out, int out_size, void* d_ws, size_t ws_size,
                              hipStream_t stream) {
    (void)in_sizes; (void)n_in; (void)out_size; (void)ws_size;
    const size_t B = BQ, L = LQ, D = DQ, NV = NVQ, BL = B * L;

    const int*   seq       = (const int*)d_in[0];
    const float* item_emb  = (const float*)d_in[1];
    const float* pos_emb   = (const float*)d_in[2];
    const float* gru_wih   = (const float*)d_in[3];
    const float* gru_whh   = (const float*)d_in[4];
    const float* gru_bih   = (const float*)d_in[5];
    const float* gru_bhh   = (const float*)d_in[6];
    const float* in_proj_w = (const float*)d_in[7];
    const float* in_proj_b = (const float*)d_in[8];
    const float* out_proj_w= (const float*)d_in[9];
    const float* out_proj_b= (const float*)d_in[10];
    const float* ln1_g     = (const float*)d_in[11];
    const float* ln1_b     = (const float*)d_in[12];
    const float* ln2_g     = (const float*)d_in[13];
    const float* ln2_b     = (const float*)d_in[14];
    const float* ff1_w     = (const float*)d_in[15];
    const float* ff1_b     = (const float*)d_in[16];
    const float* ff2_w     = (const float*)d_in[17];
    const float* ff2_b     = (const float*)d_in[18];
    float* logits          = (float*)d_out;

    char* base = (char*)d_ws;
    size_t off = 0;
    auto alloc = [&](size_t bytes) -> void* {
        off = (off + 255) & ~(size_t)255;
        void* p = base + off;
        off += bytes;
        return p;
    };

    bf16t* item_bf  = (bf16t*)alloc(NV * D * 2);
    bf16t* wih_bf   = (bf16t*)alloc(384 * 128 * 2);
    bf16t* whh_bf   = (bf16t*)alloc(384 * 128 * 2);
    bf16t* inw_bf   = (bf16t*)alloc(384 * 128 * 2);
    bf16t* outw_bf  = (bf16t*)alloc(128 * 128 * 2);
    bf16t* ff1w_bf  = (bf16t*)alloc(512 * 128 * 2);
    bf16t* ff2w_bf  = (bf16t*)alloc(128 * 512 * 2);
    float* short_u  = (float*)alloc(B * D * 4);
    bf16t* x_bf     = (bf16t*)alloc(BL * D * 2);
    float* gi       = (float*)alloc(BL * 384 * 4);   // reused as qkv after GRU
    float* h_seq    = (float*)alloc(BL * D * 4);
    bf16t* hpos_bf  = (bf16t*)alloc(BL * D * 2);
    bf16t* o_bf     = (bf16t*)alloc(BL * D * 2);
    float* attnproj = (float*)alloc(BL * D * 4);
    float* h1       = (float*)alloc(BL * D * 4);
    bf16t* h1_bf    = (bf16t*)alloc(BL * D * 2);
    bf16t* ffmid_bf = (bf16t*)alloc(BL * 512 * 2);
    float* ffout    = (float*)alloc(BL * D * 4);
    float* h_attn   = (float*)alloc(BL * D * 4);
    bf16t* ehat_bf  = (bf16t*)alloc(B * D * 2);
    float* qkv = gi;

    auto cvt = [&](const float* src, bf16t* dst, size_t n) {
        k_f32_to_bf16<<<dim3((unsigned)((n + 255) / 256)), dim3(256), 0, stream>>>(src, dst, n);
    };
    cvt(item_emb,   item_bf, NV * D);
    cvt(gru_wih,    wih_bf,  384 * 128);
    cvt(gru_whh,    whh_bf,  384 * 128);
    cvt(in_proj_w,  inw_bf,  384 * 128);
    cvt(out_proj_w, outw_bf, 128 * 128);
    cvt(ff1_w,      ff1w_bf, 512 * 128);
    cvt(ff2_w,      ff2w_bf, 128 * 512);

    // session-graph GNN representation
    k_gnn<<<dim3(BQ), dim3(128), 0, stream>>>(seq, item_emb, short_u);

    // masked embedding gather -> bf16
    {
        size_t n = BL * D;
        k_gather_x<<<dim3((unsigned)((n + 255) / 256)), dim3(256), 0, stream>>>(seq, item_emb, x_bf);
    }

    // gi = x @ Wih^T + bih   (M=BL, N=384, K=128)
    k_gemm_bf16<<<dim3(3, (unsigned)(BL / 16)), dim3(256), 0, stream>>>(
        x_bf, wih_bf, gru_bih, gi, (bf16t*)nullptr, (int)BL, 384, 128, 0);

    // GRU scan (WMMA per step) -> h_seq, (h+pos) bf16
    k_gru_scan<<<dim3((unsigned)(B / 16)), dim3(256), 0, stream>>>(
        gi, whh_bf, gru_bhh, pos_emb, h_seq, hpos_bf);

    // qkv = (h+pos) @ in_proj^T + b   (aliases gi)
    k_gemm_bf16<<<dim3(3, (unsigned)(BL / 16)), dim3(256), 0, stream>>>(
        hpos_bf, inw_bf, in_proj_b, qkv, (bf16t*)nullptr, (int)BL, 384, 128, 0);

    // attention per (batch, head)
    k_attention<<<dim3(BQ, 2), dim3(128), 0, stream>>>(qkv, o_bf);

    // out-proj: attnproj = o @ out_proj^T + b  (N=128)
    k_gemm_bf16<<<dim3(1, (unsigned)(BL / 16)), dim3(256), 0, stream>>>(
        o_bf, outw_bf, out_proj_b, attnproj, (bf16t*)nullptr, (int)BL, 128, 128, 0);

    // h1 = LN(h_seq + pos + attnproj)
    k_add_ln<<<dim3((unsigned)BL), dim3(128), 0, stream>>>(
        h_seq, attnproj, pos_emb, ln1_g, ln1_b, h1, h1_bf);

    // ffmid = relu(h1 @ ff1^T + b)  (N=512, bf16 out)
    k_gemm_bf16<<<dim3(4, (unsigned)(BL / 16)), dim3(256), 0, stream>>>(
        h1_bf, ff1w_bf, ff1_b, (float*)nullptr, ffmid_bf, (int)BL, 512, 128, 1);

    // ffout = ffmid @ ff2^T + b  (N=128, K=512)
    k_gemm_bf16<<<dim3(1, (unsigned)(BL / 16)), dim3(256), 0, stream>>>(
        ffmid_bf, ff2w_bf, ff2_b, ffout, (bf16t*)nullptr, (int)BL, 128, 512, 0);

    // h_attn = LN(h1 + ffout)
    k_add_ln<<<dim3((unsigned)BL), dim3(128), 0, stream>>>(
        h1, ffout, (const float*)nullptr, ln2_g, ln2_b, h_attn, (bf16t*)nullptr);

    // e_hat = short_u + mean(h_seq) + mean(h_attn)  -> bf16
    k_pool<<<dim3(BQ), dim3(128), 0, stream>>>(seq, h_seq, h_attn, short_u, ehat_bf);

    // logits = e_hat @ item_emb^T   (M=2048, N=50001, K=128; ragged N tile)
    {
        const int ntiles = (NVQ + 15) / 16;           // 3126
        const int gx = (ntiles + 7) / 8;              // 391
        k_gemm_bf16<<<dim3(gx, (unsigned)(B / 16)), dim3(256), 0, stream>>>(
            ehat_bf, item_bf, (const float*)nullptr, logits, (bf16t*)nullptr,
            (int)B, NVQ, 128, 0);
    }
}
